// spk_vq_vae_3L_50294067036779
// MI455X (gfx1250) — compile-verified, matrix operand order checked by reading, not run on hardware
//
#include <hip/hip_runtime.h>
#include <hip/hip_bf16.h>
#include <math.h>

typedef _Float16 half_t;
typedef __attribute__((ext_vector_type(16))) _Float16 v16h;
typedef __attribute__((ext_vector_type(8)))  _Float16 v8h;
typedef __attribute__((ext_vector_type(8)))  float    v8f;

union HalfPack { v16h v; v8h h[2]; };

#define BN_EPS 1e-5f

// ---------------------------------------------------------------------------
// Core: implicit-conv GEMM with WMMA f32_16x16x32_f16, with optional fused
// per-channel sum/sumsq accumulation (training-mode BatchNorm statistics).
// X: [M, Ks] f16 rows m = b*L + l (Ks multiple of 32, zero padded)
// Wp: [taps, Npad, Ks] f16 (Bt layout: row n holds K-contiguous weights)
// Y: [M, Npad] f32.  taps=1,pad=0 => plain GEMM (pass L=M).
// stats: null, or float[2*Npad] zero-initialized; receives column sums/sumsq.
// NT = number of 16-wide N tiles per wave (compile-time): every launch in this
// net has exactly Npad == gridDim.y * NT * 16, so all tiles are unconditionally
// valid -> NT independent back-to-back v_wmma per 32-K chunk, no branches.
// All launches use M % 128 == 0, so every wave is active (convergent syncs).
// ---------------------------------------------------------------------------
template <int NT>
__global__ __launch_bounds__(256) void k_conv_wmma(
    const half_t* __restrict__ X, const half_t* __restrict__ Wp,
    const float* __restrict__ bias, int bias_n,
    float* __restrict__ Y, float* __restrict__ stats,
    int M, int L, int Npad, int Ks, int taps, int pad)
{
  __shared__ float sst[512];              // 2 * max Npad(256) partial stats
  const int wave = threadIdx.x >> 5;
  const int lane = threadIdx.x & 31;
  const int m0 = blockIdx.x * 128 + wave * 16;
  const int n_base = blockIdx.y * (NT * 16);

  if (stats != nullptr) {
    for (int i = threadIdx.x; i < 2 * Npad; i += 256) sst[i] = 0.f;
    __syncthreads();
  }

  const int koff = (lane >> 4) * 8;       // A/B element->K mapping (16-bit layout)
  const int m = m0 + (lane & 15);
  const int l = m % L;

  v8f acc[NT] = {};

  for (int t = 0; t < taps; ++t) {
    const int shift = t - pad;
    const int lp = l + shift;
    const bool valid = (lp >= 0) && (lp < L);
    const half_t* xrow = X + (size_t)(m + shift) * (size_t)Ks;
    const half_t* wt = Wp + (size_t)t * (size_t)Npad * (size_t)Ks
                     + (size_t)(n_base + (lane & 15)) * (size_t)Ks;
    for (int kc = 0; kc < Ks; kc += 32) {
      v16h afrag = {};
      if (valid) {
        HalfPack ua;
        ua.h[0] = *(const v8h*)(xrow + kc + koff);        // K = kc+koff .. +7
        ua.h[1] = *(const v8h*)(xrow + kc + koff + 16);   // K = kc+koff+16 .. +23
        afrag = ua.v;
        if (kc + 32 < Ks) __builtin_prefetch(xrow + kc + 32 + koff, 0, 1);
      }
#pragma unroll
      for (int nt = 0; nt < NT; ++nt) {
        const half_t* wrow = wt + (size_t)(nt * 16) * (size_t)Ks;
        HalfPack ub;
        ub.h[0] = *(const v8h*)(wrow + kc + koff);
        ub.h[1] = *(const v8h*)(wrow + kc + koff + 16);
        acc[nt] = __builtin_amdgcn_wmma_f32_16x16x32_f16(
            false, afrag, false, ub.v, (short)0, acc[nt], false, false);
      }
    }
  }

  // D layout: col = n_tile + (lane&15); rows r + 8*(lane>=16)
  const int rbase = m0 + ((lane >> 4) << 3);
#pragma unroll
  for (int nt = 0; nt < NT; ++nt) {
    const int n = n_base + nt * 16 + (lane & 15);
    float bs = 0.f;
    if (bias != nullptr && n < bias_n) bs = bias[n];
    float s = 0.f, q = 0.f;
#pragma unroll
    for (int r = 0; r < 8; ++r) {
      const float v = acc[nt][r] + bs;
      Y[(size_t)(rbase + r) * (size_t)Npad + n] = v;
      s += v; q += v * v;
    }
    if (stats != nullptr) {
      atomicAdd(&sst[n], s);              // ds_add_f32, 16 contributors/column
      atomicAdd(&sst[Npad + n], q);
    }
  }

  if (stats != nullptr) {
    __syncthreads();
    for (int i = threadIdx.x; i < 2 * Npad; i += 256)
      atomicAdd(&stats[i], sst[i]);
  }
}

// Normalize (+ReLU or +maxpool over adjacent row pairs), f16 out.
__global__ __launch_bounds__(256) void k_bn_apply(
    const float* __restrict__ Y, half_t* __restrict__ out,
    const float* __restrict__ stats, const float* __restrict__ gamma,
    const float* __restrict__ beta, int N, float invM,
    int do_relu, int do_pool, long n)
{
  long i = (long)blockIdx.x * 256 + threadIdx.x;
  if (i >= n) return;
  const int c = (int)(i % N);
  const long m = i / N;
  float mean = stats[c] * invM;
  float var  = stats[N + c] * invM - mean * mean;
  float sc = gamma[c] * rsqrtf(var + BN_EPS);
  float sh = beta[c] - mean * sc;
  float v;
  if (do_pool) {
    float a = Y[(size_t)(2 * m) * N + c] * sc + sh;
    float b = Y[(size_t)(2 * m + 1) * N + c] * sc + sh;
    v = fmaxf(a, b);
  } else {
    v = Y[(size_t)m * N + c] * sc + sh;
    if (do_relu) v = fmaxf(v, 0.f);
  }
  out[i] = (half_t)v;
}

__global__ __launch_bounds__(256) void k_zero_f32(float* __restrict__ p, long n) {
  long i = (long)blockIdx.x * 256 + threadIdx.x;
  if (i < n) p[i] = 0.f;
}

// x [B,4,64] f32 (NCL) -> X0 [B*64, 32] f16 (cols >=4 zero)
__global__ __launch_bounds__(256) void k_pack_input(
    const float* __restrict__ x, half_t* __restrict__ out, long n) {
  long i = (long)blockIdx.x * 256 + threadIdx.x;
  if (i >= n) return;
  const int k = (int)(i & 31);
  const long m = i >> 5;
  const long b = m >> 6;
  const int l = (int)(m & 63);
  float v = (k < 4) ? x[b * 256 + (long)k * 64 + l] : 0.f;
  out[i] = (half_t)v;
}

// Pack conv weights to [taps, Npad, Ks] f16.
// enc (dec=0): w[O][I][T];  dec (dec=1): w[I][O][T], flipped+transposed.
__global__ __launch_bounds__(256) void k_pack_w(
    const float* __restrict__ w, half_t* __restrict__ out,
    int O, int I, int T, int Npad, int Ks, int dec, long n) {
  long i = (long)blockIdx.x * 256 + threadIdx.x;
  if (i >= n) return;
  const int t = (int)(i / ((long)Npad * Ks));
  const int r = (int)(i % ((long)Npad * Ks));
  const int o = r / Ks;
  const int k = r % Ks;
  float v = 0.f;
  if (o < O && k < I)
    v = dec ? w[(long)k * O * T + (long)o * T + (T - 1 - t)]
            : w[(long)o * I * T + (long)k * T + t];
  out[i] = (half_t)v;
}

// enc output [B*8, 256] f16 -> Xflat [B, 2048] f16 (q = c*8 + l)
__global__ __launch_bounds__(256) void k_repack_fc1(
    const half_t* __restrict__ act, half_t* __restrict__ dst, long n) {
  long i = (long)blockIdx.x * 256 + threadIdx.x;
  if (i >= n) return;
  const long b = i >> 11;
  const int q = (int)(i & 2047);
  const int c = q >> 3, l = q & 7;
  dst[i] = act[(b * 8 + l) * 256 + c];
}

__global__ __launch_bounds__(256) void k_tanh_split(
    const float* __restrict__ Y, float* __restrict__ zf,
    half_t* __restrict__ zh, long n) {
  long i = (long)blockIdx.x * 256 + threadIdx.x;
  if (i >= n) return;
  float v = tanhf(Y[i]);
  zf[i] = v; zh[i] = (half_t)v;
}

__global__ __launch_bounds__(256) void k_cast_f16(
    const float* __restrict__ s, half_t* __restrict__ d, long n) {
  long i = (long)blockIdx.x * 256 + threadIdx.x;
  if (i < n) d[i] = (half_t)s[i];
}

// dst [N, K] f16 = transpose of src [K, N] f32
__global__ __launch_bounds__(256) void k_transpose_f16(
    const float* __restrict__ s, half_t* __restrict__ d, int K, int N, long n) {
  long i = (long)blockIdx.x * 256 + threadIdx.x;
  if (i >= n) return;
  const long nn = i / K;
  const int k = (int)(i % K);
  d[i] = (half_t)s[(long)k * N + nn];
}

__global__ __launch_bounds__(256) void k_row_norms(
    const float* __restrict__ E, float* __restrict__ norms, int J, int D) {
  int j = blockIdx.x * 256 + threadIdx.x;
  if (j >= J) return;
  float s = 0.f;
  for (int d = 0; d < D; ++d) { float v = E[(long)j * D + d]; s += v * v; }
  norms[j] = s;
}

__global__ __launch_bounds__(256) void k_vq_argmin(
    const float* __restrict__ S, const float* __restrict__ norms,
    int* __restrict__ jout, int J) {
  const int b = blockIdx.x;
  const float* row = S + (size_t)b * J;
  float best = 3.4e38f; int bj = J;
  for (int j = threadIdx.x; j < J; j += 256) {
    float d = norms[j] - 2.0f * row[j];
    if (d < best || (d == best && j < bj)) { best = d; bj = j; }
  }
  __shared__ float bd[256]; __shared__ int bi[256];
  bd[threadIdx.x] = best; bi[threadIdx.x] = bj;
  __syncthreads();
  for (int off = 128; off > 0; off >>= 1) {
    if (threadIdx.x < off) {
      float od = bd[threadIdx.x + off]; int oj = bi[threadIdx.x + off];
      if (od < bd[threadIdx.x] || (od == bd[threadIdx.x] && oj < bi[threadIdx.x])) {
        bd[threadIdx.x] = od; bi[threadIdx.x] = oj;
      }
    }
    __syncthreads();
  }
  if (threadIdx.x == 0) jout[b] = bi[0];
}

__global__ __launch_bounds__(128) void k_vq_loss(
    const float* __restrict__ z, const float* __restrict__ E,
    const int* __restrict__ j, float* __restrict__ out2, int B, int D) {
  const int b = blockIdx.x;
  const int d = threadIdx.x;
  float diff = z[(long)b * D + d] - E[(long)j[b] * D + d];
  __shared__ float sm[128];
  sm[d] = diff * diff;
  __syncthreads();
  for (int off = 64; off > 0; off >>= 1) {
    if (d < off) sm[d] += sm[d + off];
    __syncthreads();
  }
  if (d == 0) {
    float r = sm[0] / (float)B;
    atomicAdd(&out2[0], r);
    atomicAdd(&out2[1], r);
  }
}

__global__ __launch_bounds__(256) void k_gather_codes(
    const float* __restrict__ E, const int* __restrict__ j,
    half_t* __restrict__ W, long n) {
  long i = (long)blockIdx.x * 256 + threadIdx.x;
  if (i >= n) return;
  const long b = i >> 7;
  const int d = (int)(i & 127);
  W[i] = (half_t)E[(long)j[b] * 128 + d];
}

// fc2 out [B,2048] f32 -> tanh -> act [B*8, 256] f16 (q = c*8 + l)
__global__ __launch_bounds__(256) void k_tanh_unpack_dec(
    const float* __restrict__ Y, half_t* __restrict__ act, long n) {
  long i = (long)blockIdx.x * 256 + threadIdx.x;
  if (i >= n) return;
  const long b = i >> 11;
  const int r = (int)(i & 2047);
  const int l = r >> 8, c = r & 255;
  act[i] = (half_t)tanhf(Y[b * 2048 + (long)c * 8 + l]);
}

// nearest upsample x2 along L: out row m2 <- in row m2/2
__global__ __launch_bounds__(256) void k_upsample2(
    const half_t* __restrict__ in, half_t* __restrict__ out, int C, long n) {
  long i = (long)blockIdx.x * 256 + threadIdx.x;
  if (i >= n) return;
  const long m2 = i / C;
  const int c = (int)(i % C);
  out[i] = in[(m2 >> 1) * C + c];
}

// Y [B*64, 16] f32 -> d_out [B,4,64] f32 NCL
__global__ __launch_bounds__(256) void k_finalize(
    const float* __restrict__ Y, float* __restrict__ out, long n) {
  long i = (long)blockIdx.x * 256 + threadIdx.x;
  if (i >= n) return;
  const long b = i >> 8;
  const int r = (int)(i & 255);
  const int c = r >> 6, l = r & 63;
  out[i] = Y[(b * 64 + l) * 16 + c];
}

// ---------------------------------------------------------------------------
extern "C" void kernel_launch(void* const* d_in, const int* in_sizes, int n_in,
                              void* d_out, int out_size, void* d_ws, size_t ws_size,
                              hipStream_t stream) {
  (void)in_sizes; (void)n_in; (void)out_size; (void)ws_size;
  const int B = 8192;
  const long RECON = (long)B * 4 * 64;

  const float* x = (const float*)d_in[0];
  const float* enc_w[8], *enc_b[8], *enc_g[8], *enc_be[8];
  for (int i = 0; i < 8; ++i) {
    enc_w[i]  = (const float*)d_in[1 + i];
    enc_b[i]  = (const float*)d_in[9 + i];
    enc_g[i]  = (const float*)d_in[17 + i];
    enc_be[i] = (const float*)d_in[25 + i];
  }
  const float* fc1_w = (const float*)d_in[33];
  const float* fc1_b = (const float*)d_in[34];
  const float* fc2_w = (const float*)d_in[35];
  const float* fc2_b = (const float*)d_in[36];
  const float* dec_w[3], *dec_b[3];
  for (int i = 0; i < 3; ++i) { dec_w[i] = (const float*)d_in[37 + i]; dec_b[i] = (const float*)d_in[40 + i]; }
  const float* dec_g[2]  = { (const float*)d_in[43], (const float*)d_in[44] };
  const float* dec_be[2] = { (const float*)d_in[45], (const float*)d_in[46] };
  const float* embed = (const float*)d_in[47];
  float* out = (float*)d_out;

  // --- workspace carve ---
  char* p = (char*)d_ws;
  auto carve = [&](size_t bytes) -> void* {
    void* r = (void*)p; p += (bytes + 255) & ~(size_t)255; return r;
  };
  const long ACT = 33554432;                      // max f16 activation elems
  half_t* actA  = (half_t*)carve((size_t)ACT * 2);
  half_t* actB  = (half_t*)carve((size_t)ACT * 2);
  float*  Ybuf  = (float*)carve((size_t)ACT * 4); // max f32 GEMM-out elems
  half_t* wpack = (half_t*)carve(262144 * 2);
  half_t* fc1Bt = (half_t*)carve((size_t)128 * 2048 * 2);
  half_t* fc2Bt = (half_t*)carve((size_t)2048 * 128 * 2);
  half_t* embH  = (half_t*)carve((size_t)1024 * 128 * 2);
  half_t* wselH = (half_t*)carve((size_t)B * 128 * 2);
  float*  zf32  = (float*)carve((size_t)B * 128 * 4);
  half_t* zh16  = (half_t*)carve((size_t)B * 128 * 2);
  float*  stats = (float*)carve(512 * 4);
  float*  norms = (float*)carve(1024 * 4);
  int*    jbuf  = (int*)carve((size_t)B * 4);

  auto ew = [](long n) { return dim3((unsigned)((n + 255) / 256)); };
  const dim3 T256(256), T128(128);

  // conv/GEMM launcher: NT=4 when Npad is a multiple of 64, else NT=1 (Npad=16)
  auto conv = [&](const half_t* X, const half_t* Wp, const float* bias, int bias_n,
                  float* Y, float* st, int M, int L, int Npad, int Ks, int taps, int pad) {
    if (Npad % 64 == 0) {
      dim3 g(M / 128, Npad / 64);
      k_conv_wmma<4><<<g, T256, 0, stream>>>(X, Wp, bias, bias_n, Y, st,
                                             M, L, Npad, Ks, taps, pad);
    } else {
      dim3 g(M / 128, Npad / 16);
      k_conv_wmma<1><<<g, T256, 0, stream>>>(X, Wp, bias, bias_n, Y, st,
                                             M, L, Npad, Ks, taps, pad);
    }
  };

  // --- one-time-per-call weight packing ---
  k_transpose_f16<<<ew(128L * 2048), T256, 0, stream>>>(fc1_w, fc1Bt, 2048, 128, 128L * 2048);
  k_transpose_f16<<<ew(2048L * 128), T256, 0, stream>>>(fc2_w, fc2Bt, 128, 2048, 2048L * 128);
  k_cast_f16<<<ew(1024L * 128), T256, 0, stream>>>(embed, embH, 1024L * 128);
  k_row_norms<<<dim3(4), T256, 0, stream>>>(embed, norms, 1024, 128);

  // --- input pack ---
  k_pack_input<<<ew((long)B * 64 * 32), T256, 0, stream>>>(x, actA, (long)B * 64 * 32);

  // --- encoder ---
  struct EncCfg { int O, I, Ks, T, pad, L, relu, pool; };
  const EncCfg enc[8] = {
    { 64,   4,  32, 5, 2, 64, 1, 0}, { 64,  64,  64, 5, 2, 64, 0, 1},
    {128,  64,  64, 3, 1, 32, 1, 0}, {128, 128, 128, 3, 1, 32, 1, 0},
    {128, 128, 128, 3, 1, 32, 0, 1}, {256, 128, 128, 3, 1, 16, 1, 0},
    {256, 256, 256, 3, 1, 16, 1, 0}, {256, 256, 256, 3, 1, 16, 0, 1},
  };
  half_t* cur = actA; half_t* nxt = actB;
  for (int li = 0; li < 8; ++li) {
    const EncCfg& c = enc[li];
    const int M = B * c.L;
    const long wn = (long)c.T * c.O * c.Ks;
    k_pack_w<<<ew(wn), T256, 0, stream>>>(enc_w[li], wpack, c.O, c.I, c.T, c.O, c.Ks, 0, wn);
    k_zero_f32<<<ew(2L * c.O), T256, 0, stream>>>(stats, 2L * c.O);
    conv(cur, wpack, enc_b[li], c.O, Ybuf, stats, M, c.L, c.O, c.Ks, c.T, c.pad);
    const long Mout = c.pool ? (long)M / 2 : (long)M;
    k_bn_apply<<<ew(Mout * c.O), T256, 0, stream>>>(
        Ybuf, nxt, stats, enc_g[li], enc_be[li], c.O, 1.0f / (float)M, c.relu, c.pool, Mout * c.O);
    half_t* t = cur; cur = nxt; nxt = t;
  }
  // encoder output: cur = [B*8, 256] f16

  // --- FC1 + tanh ---
  k_repack_fc1<<<ew((long)B * 2048), T256, 0, stream>>>(cur, nxt, (long)B * 2048);
  conv(nxt, fc1Bt, fc1_b, 128, Ybuf, nullptr, B, B, 128, 2048, 1, 0);
  k_tanh_split<<<ew((long)B * 128), T256, 0, stream>>>(Ybuf, zf32, zh16, (long)B * 128);

  // --- VQ: scores = z_e @ embed^T, argmin(norms - 2*scores) ---
  conv(zh16, embH, nullptr, 0, Ybuf, nullptr, B, B, 1024, 128, 1, 0);
  k_vq_argmin<<<dim3(B), T256, 0, stream>>>(Ybuf, norms, jbuf, 1024);
  k_zero_f32<<<dim3(1), T256, 0, stream>>>(out + RECON, 2);
  k_vq_loss<<<dim3(B), T128, 0, stream>>>(zf32, embed, jbuf, out + RECON, B, 128);
  k_gather_codes<<<ew((long)B * 128), T256, 0, stream>>>(embed, jbuf, wselH, (long)B * 128);

  // --- FC2 + tanh -> [B*8, 256] f16 in actA ---
  conv(wselH, fc2Bt, fc2_b, 2048, Ybuf, nullptr, B, B, 2048, 128, 1, 0);
  k_tanh_unpack_dec<<<ew((long)B * 2048), T256, 0, stream>>>(Ybuf, actA, (long)B * 2048);

  // --- decoder stages 0,1: upsample -> deconv -> BN -> ReLU ---
  struct DecCfg { int I, O, T, pad, Lout; };
  const DecCfg dec[2] = { {256, 128, 3, 1, 16}, {128, 64, 3, 1, 32} };
  for (int s = 0; s < 2; ++s) {
    const DecCfg& c = dec[s];
    const int M = B * c.Lout;
    k_upsample2<<<ew((long)M * c.I), T256, 0, stream>>>(actA, actB, c.I, (long)M * c.I);
    const long wn = (long)c.T * c.O * c.I;
    k_pack_w<<<ew(wn), T256, 0, stream>>>(dec_w[s], wpack, c.O, c.I, c.T, c.O, c.I, 1, wn);
    k_zero_f32<<<ew(2L * c.O), T256, 0, stream>>>(stats, 2L * c.O);
    conv(actB, wpack, dec_b[s], c.O, Ybuf, stats, M, c.Lout, c.O, c.I, c.T, c.pad);
    k_bn_apply<<<ew((long)M * c.O), T256, 0, stream>>>(
        Ybuf, actA, stats, dec_g[s], dec_be[s], c.O, 1.0f / (float)M, 1, 0, (long)M * c.O);
  }

  // --- final: upsample -> deconv(64->4, K5, pad'=2) + bias -> d_out ---
  const int Mf = B * 64;
  k_upsample2<<<ew((long)Mf * 64), T256, 0, stream>>>(actA, actB, 64, (long)Mf * 64);
  const long wn = 5L * 16 * 64;
  k_pack_w<<<ew(wn), T256, 0, stream>>>(dec_w[2], wpack, 4, 64, 5, 16, 64, 1, wn);
  conv(actB, wpack, dec_b[2], 4, Ybuf, nullptr, Mf, 64, 16, 64, 5, 2);
  k_finalize<<<ew(RECON), T256, 0, stream>>>(Ybuf, out, RECON);
}
